// SuperPoint_27161373179985
// MI455X (gfx1250) — matrix operand order repaired and checked by reading, not compiled, not run on hardware
//
#include <hip/hip_runtime.h>
#include <hip/hip_bf16.h>

typedef _Float16 v16h __attribute__((ext_vector_type(16)));
typedef _Float16 v8h  __attribute__((ext_vector_type(8)));
typedef float    v8f  __attribute__((ext_vector_type(8)));
typedef unsigned int u32x4 __attribute__((ext_vector_type(4)));
typedef int i32x4 __attribute__((ext_vector_type(4)));
typedef int i32x8 __attribute__((ext_vector_type(8)));

#define KP   1024
#define MAXC 32768
#define CDIV(a,b) (((a)+(b)-1)/(b))

// ============================ prep kernels ============================
__global__ void sp_cast_f16(const float* __restrict__ x, _Float16* __restrict__ y, int n) {
  int t = blockIdx.x * blockDim.x + threadIdx.x;
  if (t < n) y[t] = (_Float16)x[t];
}

// zero the 1-pixel halo ring of a padded NHWC tensor [2][H+2][W+2][C]
__global__ void sp_zero_halo(_Float16* __restrict__ X, int H, int W, int C) {
  int perim = 2 * (W + 2) + 2 * H;
  int n = 2 * perim * C;
  int t = blockIdx.x * blockDim.x + threadIdx.x;
  if (t >= n) return;
  int c = t % C, r = t / C;
  int b = r / perim, i = r - b * perim;
  int y, x;
  if (i < W + 2)            { y = 0;     x = i; }
  else if (i < 2 * (W + 2)) { y = H + 1; x = i - (W + 2); }
  else { int j = i - 2 * (W + 2); y = 1 + (j >> 1); x = (j & 1) ? (W + 1) : 0; }
  X[(((size_t)(b * (H + 2) + y)) * (W + 2) + x) * C + c] = (_Float16)0.f;
}

// pack weights [Cout,Cin,ksz,ksz] (fp32) -> [Cpad][Kpad] f16 with K = tap*Cin + ci
__global__ void sp_pack_w(const float* __restrict__ w, _Float16* __restrict__ o,
                          int Cout, int Cin, int KK, int Kpad, int n) {
  int t = blockIdx.x * blockDim.x + threadIdx.x;
  if (t >= n) return;
  int co = t / Kpad, k = t - co * Kpad;
  int tap = k / Cin, ci = k - tap * Cin;
  float v = (co < Cout && tap < KK) ? w[(size_t)(co * Cin + ci) * KK + tap] : 0.f;
  o[t] = (_Float16)v;
}

__global__ void sp_prep_p(const float* __restrict__ b, const float* __restrict__ g,
                          const float* __restrict__ beta, const float* __restrict__ mu,
                          const float* __restrict__ var, float* __restrict__ P,
                          int Cout, int Cpad) {
  int co = blockIdx.x * blockDim.x + threadIdx.x;
  if (co >= Cpad) return;
  float bb = 0.f, sc = 0.f, sh = 0.f;
  if (co < Cout) {
    float inv = rsqrtf(var[co] + 1e-5f);
    sc = g[co] * inv;
    bb = b[co];
    sh = beta[co] - mu[co] * sc;
  }
  P[co * 3 + 0] = bb; P[co * 3 + 1] = sc; P[co * 3 + 2] = sh;
}

// ================= first layer (Cin=1): tiny direct conv, padded output =================
__global__ void sp_conv_c1(const _Float16* __restrict__ X, const float* __restrict__ w,
                           const float* __restrict__ P, _Float16* __restrict__ Y,
                           int H, int W, int Cout) {
  int HW = H * W;
  int t = blockIdx.x * blockDim.x + threadIdx.x;
  if (t >= 2 * HW) return;
  int b = t / HW, p = t - b * HW;
  int py = p / W, px = p - py * W;
  float xin[9];
#pragma unroll
  for (int r = 0; r < 9; ++r) {
    int dy = r / 3 - 1, dx = r % 3 - 1;
    int yy = py + dy, xx = px + dx;
    xin[r] = (yy >= 0 && yy < H && xx >= 0 && xx < W)
                 ? (float)X[(size_t)b * HW + (size_t)yy * W + xx] : 0.f;
  }
  _Float16* dst = Y + (((size_t)(b * (H + 2) + py + 1)) * (W + 2) + px + 1) * Cout;
  for (int co = 0; co < Cout; ++co) {
    float s = P[co * 3 + 0];
#pragma unroll
    for (int r = 0; r < 9; ++r) s += w[co * 9 + r] * xin[r];
    s = fmaxf(s, 0.f);
    s = s * P[co * 3 + 1] + P[co * 3 + 2];
    dst[co] = (_Float16)s;
  }
}

// ======================= WMMA implicit-GEMM conv (padded NHWC) =======================
// X: [2,H+2,W+2,Cin] f16 with zeroed halo, Cin = 1<<lc. Wt: [Cpad][Kpad], K = tap*Cin+ci.
// Weight tile DMA'd to LDS via TDM; branch-free inner loop:
//   2x ds_load_b128 (A) + 2x global_load_b128 (B) + v_wmma_f32_16x16x32_f16.
template <int KSZ>
__global__ void sp_conv(const _Float16* __restrict__ X,
                        const _Float16* __restrict__ Wt,
                        const float*    __restrict__ P,
                        _Float16*       __restrict__ Y,
                        int lc, int H, int W, int Cstore, int Kpad, int relu, int padout) {
  extern __shared__ _Float16 ldsw[];          // [16][Kpad] weight tile
  const int lane = threadIdx.x & 31;
  const int wave = threadIdx.x >> 5;
  const int HW = H * W;
  const int co0 = blockIdx.y * 16;
  const int b = blockIdx.z;
  const int Cin = 1 << lc;

#if defined(__gfx1250__) && __has_builtin(__builtin_amdgcn_tensor_load_to_lds)
  if (wave == 0) {
    // 2D tensor DMA descriptor (CDNA5 ISA)
    unsigned long long ga = (unsigned long long)(const void*)(Wt + (size_t)co0 * Kpad);
    unsigned int la = (unsigned int)(unsigned long long)(const void*)&ldsw[0];
    unsigned Cpad = gridDim.y * 16u, Kp = (unsigned)Kpad;
    u32x4 g0;
    g0[0] = 1u;                                            // count=1 (valid user D#)
    g0[1] = la;                                            // lds_addr (bytes)
    g0[2] = (unsigned)(ga & 0xffffffffu);                  // global_addr[31:0]
    g0[3] = (unsigned)((ga >> 32) & 0x01ffffffu) | 0x80000000u; // addr[56:32] | type=2
    i32x8 g1;
    g1[0] = (int)(1u << 16);                               // data_size=1 (2B); mask=0
    g1[1] = (int)(Kp << 16);                               // tensor_dim0[15:0]
    g1[2] = (int)((Kp >> 16) | (Cpad << 16));              // dim0[31:16] | dim1[15:0]
    g1[3] = (int)((Cpad >> 16) | (Kp << 16));              // dim1[31:16] | tile_dim0
    g1[4] = 16;                                            // tile_dim1=16, tile_dim2=0
    g1[5] = (int)Kp;                                       // tensor_dim0_stride[31:0]
    g1[6] = 0; g1[7] = 0;
    i32x4 z4 = {0, 0, 0, 0};
#if __clang_major__ >= 23
    i32x8 z8 = {0, 0, 0, 0, 0, 0, 0, 0};
    __builtin_amdgcn_tensor_load_to_lds(g0, g1, z4, z4, z8, 0);
#else
    __builtin_amdgcn_tensor_load_to_lds(g0, g1, z4, z4, 0);
#endif
    __builtin_amdgcn_s_wait_tensorcnt((short)0);
  }
  __syncthreads();
#else
  for (int i = threadIdx.x; i < 16 * Kpad / 8; i += blockDim.x)
    ((v8h*)ldsw)[i] = ((const v8h*)(Wt + (size_t)co0 * Kpad))[i];
  __syncthreads();
#endif

  const int p0 = (blockIdx.x * (blockDim.x >> 5) + wave) * 16;
  if (p0 >= HW) return;                 // wave-uniform: EXEC all-1s for WMMA
  const int halfsel = lane >> 4;
  const int p = p0 + (lane & 15);
  const int py = p / W, px = p - py * W;
  const int Wp = W + 2;
  const _Float16* lrow = ldsw + (size_t)(lane & 15) * Kpad;
  const _Float16* xrow = X + (size_t)b * (H + 2) * Wp * Cin;
  __builtin_prefetch(xrow + ((size_t)(py + 1) * Wp + px + 1) * Cin, 0, 1);

  v8f acc = {};
#pragma unroll
  for (int tap = 0; tap < KSZ * KSZ; ++tap) {
    const int dy = tap / KSZ - (KSZ >> 1);
    const int dx = tap % KSZ - (KSZ >> 1);
    const _Float16* bsrc = xrow + ((size_t)(py + dy + 1) * Wp + (px + dx + 1)) * Cin;
    const int kt = tap * Cin;
    for (int ci0 = 0; ci0 < Cin; ci0 += 32) {
      const int k0 = kt + ci0;
      v8h alo = *(const v8h*)(lrow + k0 + halfsel * 8);        // ds_load_b128
      v8h ahi = *(const v8h*)(lrow + k0 + 16 + halfsel * 8);
      v16h a;
#pragma unroll
      for (int i = 0; i < 8; ++i) { a[i] = alo[i]; a[8 + i] = ahi[i]; }

      const int cib = ci0 + halfsel * 8;
      v8h blo = *(const v8h*)(bsrc + cib);                     // halo => real zeros
      v8h bhi = *(const v8h*)(bsrc + cib + 16);
      v16h bf;
#pragma unroll
      for (int i = 0; i < 8; ++i) { bf[i] = blo[i]; bf[8 + i] = bhi[i]; }

      acc = __builtin_amdgcn_wmma_f32_16x16x32_f16(false, a, false, bf,
                                                   (short)0, acc, false, false);
    }
  }

  // epilogue: bias -> relu -> BN, one b128 store of 8 consecutive channels
  const int cobase = co0 + halfsel * 8;
  v8h o;
#pragma unroll
  for (int r = 0; r < 8; ++r) {
    int co = cobase + r;
    float v = acc[r] + P[co * 3 + 0];
    if (relu) v = fmaxf(v, 0.f);
    v = v * P[co * 3 + 1] + P[co * 3 + 2];
    o[r] = (_Float16)v;
  }
  size_t opix = padout ? (((size_t)(b * (H + 2) + py + 1)) * Wp + (px + 1))
                       : ((size_t)b * HW + p);
  *(v8h*)(Y + opix * Cstore + cobase) = o;
}

// ============================ pooling (padded NHWC in/out) ============================
__global__ void sp_maxpool2(const _Float16* __restrict__ X, _Float16* __restrict__ Y,
                            int H, int W, int C, int n) {
  int t = blockIdx.x * blockDim.x + threadIdx.x;
  if (t >= n) return;
  int W2 = W >> 1, H2 = H >> 1;
  int c = t % C, rest = t / C;
  int x = rest % W2; rest /= W2;
  int y = rest % H2; int b = rest / H2;
  const _Float16* base =
      X + (((size_t)(b * (H + 2) + 2 * y + 1)) * (W + 2) + 2 * x + 1) * C + c;
  size_t rs = (size_t)(W + 2) * C;
  float m = fmaxf(fmaxf((float)base[0], (float)base[C]),
                  fmaxf((float)base[rs], (float)base[rs + C]));
  Y[(((size_t)(b * (H2 + 2) + y + 1)) * (W2 + 2) + x + 1) * C + c] = (_Float16)m;
}

// ===================== softmax + pixel shuffle (NHWC, stride 80) =====================
__global__ void sp_softmax_shuffle(const _Float16* __restrict__ D, float* __restrict__ SC,
                                   int h, int w) {
  int hw = h * w;
  int t = blockIdx.x * blockDim.x + threadIdx.x;
  if (t >= 2 * hw) return;
  int b = t / hw, p = t - b * hw;
  int y = p / w, x = p - y * w;
  const _Float16* base = D + (size_t)t * 80;
  float mx = -1e30f;
  for (int c = 0; c < 65; ++c) mx = fmaxf(mx, (float)base[c]);
  float sum = 0.f;
  for (int c = 0; c < 65; ++c) sum += __expf((float)base[c] - mx);
  float inv = 1.f / sum;
  int Wf = w * 8;
  for (int c = 0; c < 64; ++c) {
    float v = __expf((float)base[c] - mx) * inv;
    int sy = c >> 3, sx = c & 7;
    SC[(size_t)b * (h * 8) * Wf + (size_t)(y * 8 + sy) * Wf + (x * 8 + sx)] = v;
  }
}

// ============================ NMS ============================
__global__ void sp_winmax_h(const float* __restrict__ in, float* __restrict__ out,
                            int W, int n, int r) {
  int t = blockIdx.x * blockDim.x + threadIdx.x;
  if (t >= n) return;
  int x = t % W;
  const float* row = in + (t - x);
  int x0 = max(x - r, 0), x1 = min(x + r, W - 1);
  float m = -1e30f;
  for (int xx = x0; xx <= x1; ++xx) m = fmaxf(m, row[xx]);
  out[t] = m;
}

__global__ void sp_winmax_v(const float* __restrict__ in, float* __restrict__ out,
                            int H, int W, int n, int r) {
  int t = blockIdx.x * blockDim.x + threadIdx.x;
  if (t >= n) return;
  int img = t / (H * W), rr = t - img * (H * W);
  int y = rr / W, x = rr - y * W;
  int y0 = max(y - r, 0), y1 = min(y + r, H - 1);
  const float* base = in + (size_t)img * H * W + x;
  float m = -1e30f;
  for (int yy = y0; yy <= y1; ++yy) m = fmaxf(m, base[(size_t)yy * W]);
  out[t] = m;
}

__global__ void sp_eqmask(const float* __restrict__ S, const float* __restrict__ MP,
                          float* __restrict__ M, int n) {
  int t = blockIdx.x * blockDim.x + threadIdx.x;
  if (t < n) M[t] = (S[t] == MP[t]) ? 1.f : 0.f;
}

__global__ void sp_suppress(const float* __restrict__ supp, const float* __restrict__ S,
                            float* __restrict__ out, int n) {
  int t = blockIdx.x * blockDim.x + threadIdx.x;
  if (t < n) out[t] = (supp[t] > 0.f) ? 0.f : S[t];
}

__global__ void sp_mask_update(float* __restrict__ M, const float* __restrict__ supp,
                               const float* __restrict__ s2, const float* __restrict__ mp2,
                               int n) {
  int t = blockIdx.x * blockDim.x + threadIdx.x;
  if (t >= n) return;
  if ((s2[t] == mp2[t]) && !(supp[t] > 0.f)) M[t] = 1.f;
}

__global__ void sp_finalize(const float* __restrict__ M, const float* __restrict__ S,
                            float* __restrict__ SF, int H, int W) {
  int n = 2 * H * W;
  int t = blockIdx.x * blockDim.x + threadIdx.x;
  if (t >= n) return;
  int rr = t % (H * W);
  int y = rr / W, x = rr - y * W;
  float v = (M[t] > 0.f) ? S[t] : 0.f;
  bool border = (y >= 4) && (y < H - 4) && (x >= 4) && (x < W - 4);
  SF[t] = (border && v > 0.01f) ? v : 0.f;
}

// ======================= top-K selection =======================
__global__ void sp_init_cand(float* __restrict__ cv, int* __restrict__ ci,
                             int* __restrict__ cnt, int n) {
  int t = blockIdx.x * blockDim.x + threadIdx.x;
  if (t < 2) cnt[t] = 0;
  if (t < n) { cv[t] = -1.f; ci[t] = 0x7fffffff; }
}

__global__ void sp_compact(const float* __restrict__ SF, float* __restrict__ cv,
                           int* __restrict__ ci, int* __restrict__ cnt, int HW) {
  int t = blockIdx.x * blockDim.x + threadIdx.x;
  if (t >= 2 * HW) return;
  float v = SF[t];
  if (v > 0.f) {
    int b = t / HW, i = t - b * HW;
    int pos = atomicAdd(&cnt[b], 1);
    if (pos < MAXC) { cv[(size_t)b * MAXC + pos] = v; ci[(size_t)b * MAXC + pos] = i; }
  }
}

// descending bitonic sort over (value desc, index asc) — deterministic total order
__global__ void sp_bitonic(float* __restrict__ val, int* __restrict__ idx) {
  float* v = val + (size_t)blockIdx.x * MAXC;
  int*  id = idx + (size_t)blockIdx.x * MAXC;
  for (int k = 2; k <= MAXC; k <<= 1) {
    for (int j = k >> 1; j > 0; j >>= 1) {
      for (int t = threadIdx.x; t < MAXC; t += blockDim.x) {
        int x = t ^ j;
        if (x > t) {
          float a = v[t], c = v[x];
          int ia = id[t], ic = id[x];
          bool after  = (a < c) || (a == c && ia > ic);
          bool before = (a > c) || (a == c && ia < ic);
          bool sw = ((t & k) == 0) ? after : before;
          if (sw) { v[t] = c; v[x] = a; id[t] = ic; id[x] = ia; }
        }
      }
      __syncthreads();
    }
  }
}

__global__ void sp_emit(const float* __restrict__ cv, const int* __restrict__ ci,
                        float* __restrict__ out, int Wf) {
  int t = blockIdx.x * blockDim.x + threadIdx.x;
  if (t >= 2 * KP) return;
  int b = t / KP, k = t - b * KP;
  float v = cv[(size_t)b * MAXC + k];
  int   i = ci[(size_t)b * MAXC + k];
  if (v < 0.f) { v = 0.f; i = 0; }
  out[(size_t)(b * KP + k) * 2 + 0] = (float)(i % Wf);
  out[(size_t)(b * KP + k) * 2 + 1] = (float)(i / Wf);
  out[2 * KP * 2 + (size_t)b * KP + k] = v;
}

// ======================= descriptor path (NHWC) =======================
__global__ void sp_desc_norm(const _Float16* __restrict__ D, float* __restrict__ DN, int n) {
  int t = blockIdx.x * blockDim.x + threadIdx.x;   // over 2*hw pixels
  if (t >= n) return;
  const _Float16* src = D + (size_t)t * 256;
  float s = 0.f;
  for (int c = 0; c < 256; ++c) { float v = (float)src[c]; s += v * v; }
  float inv = 1.f / fmaxf(sqrtf(s), 1e-12f);
  float* dst = DN + (size_t)t * 256;
  for (int c = 0; c < 256; ++c) dst[c] = (float)src[c] * inv;
}

__global__ void sp_sample(const float* __restrict__ DN, const float* __restrict__ kps,
                          float* __restrict__ od) {
  __shared__ float red[256];
  int b = blockIdx.x >> 10;
  int k = blockIdx.x & (KP - 1);
  int c = threadIdx.x;
  float kx = kps[(size_t)(b * KP + k) * 2 + 0];
  float ky = kps[(size_t)(b * KP + k) * 2 + 1];
  float px = (kx + 0.5f) * 0.125f - 0.5f;
  float py = (ky + 0.5f) * 0.125f - 0.5f;
  float x0 = floorf(px), y0 = floorf(py);
  float wx1 = px - x0, wy1 = py - y0;
  int xi = (int)x0, yi = (int)y0;
  auto g = [&](int yy, int xx) -> float {
    bool valid = (xx >= 0 && xx < 96 && yy >= 0 && yy < 96);
    int xc = min(max(xx, 0), 95), yc = min(max(yy, 0), 95);
    float v = DN[(((size_t)(b * 96 + yc) * 96) + xc) * 256 + c];
    return valid ? v : 0.f;
  };
  float v = g(yi, xi) * (1.f - wy1) * (1.f - wx1)
          + g(yi, xi + 1) * (1.f - wy1) * wx1
          + g(yi + 1, xi) * wy1 * (1.f - wx1)
          + g(yi + 1, xi + 1) * wy1 * wx1;
  red[c] = v * v;
  __syncthreads();
  for (int s = 128; s > 0; s >>= 1) {
    if (c < s) red[c] += red[c + s];
    __syncthreads();
  }
  float nrm = fmaxf(sqrtf(red[0]), 1e-12f);
  od[(size_t)(b * KP + k) * 256 + c] = v / nrm;
}

// ============================ host ============================
struct SpBlk { int cin, cout, ksz, relu; };

extern "C" void kernel_launch(void* const* d_in, const int* in_sizes, int n_in,
                              void* d_out, int out_size, void* d_ws, size_t ws_size,
                              hipStream_t stream) {
  if (n_in < 73) return;
  static const SpBlk BLKS[12] = {
    {1, 64, 3, 1},   {64, 64, 3, 1},   {64, 64, 3, 1},   {64, 64, 3, 1},
    {64, 128, 3, 1}, {128, 128, 3, 1}, {128, 128, 3, 1}, {128, 128, 3, 1},
    {128, 256, 3, 1}, {256, 65, 1, 0},                       // detector
    {128, 256, 3, 1}, {256, 256, 1, 0},                      // descriptor
  };
  int cpad[12], kpad[12];
  size_t woff[12], poff[12], wtot = 0, ptot = 0;
  for (int j = 0; j < 12; ++j) {
    int K = BLKS[j].cin * BLKS[j].ksz * BLKS[j].ksz;
    kpad[j] = ((K + 31) / 32) * 32;
    cpad[j] = ((BLKS[j].cout + 15) / 16) * 16;
    woff[j] = wtot; wtot += (size_t)cpad[j] * kpad[j];
    poff[j] = ptot; ptot += (size_t)cpad[j] * 3;
  }

  // workspace carve-out
  size_t off = 0;
  auto A = [&](size_t bytes) { size_t o = off; off = (off + bytes + 255) & ~(size_t)255; return o; };
  char* ws = (char*)d_ws;
  const size_t NS   = 2ull * 768 * 768;
  const size_t hw96 = 96 * 96;
  _Float16* WGT  = (_Float16*)(ws + A(wtot * 2));
  float*    PRM  = (float*)(ws + A(ptot * 4));
  _Float16* IMG  = (_Float16*)(ws + A(NS * 2));                      // unpadded f16 image
  size_t p768 = 2ull * 770 * 770 * 64;
  _Float16* P768a = (_Float16*)(ws + A(p768 * 2));
  _Float16* P768b = (_Float16*)(ws + A(p768 * 2));
  _Float16* DET = (_Float16*)(ws + A(2ull * hw96 * 80 * 2));
  _Float16* DSC = (_Float16*)(ws + A(2ull * hw96 * 256 * 2));
  float* S  = (float*)(ws + A(NS * 4));
  float* Mm = (float*)(ws + A(NS * 4));
  float* T1 = (float*)(ws + A(NS * 4));
  float* T2 = (float*)(ws + A(NS * 4));
  float* T3 = (float*)(ws + A(NS * 4));
  float* T4 = (float*)(ws + A(NS * 4));
  float* SF = (float*)(ws + A(NS * 4));
  float* DN = (float*)(ws + A(2ull * hw96 * 256 * 4));
  float* CV = (float*)(ws + A(2ull * MAXC * 4));
  int*   CI = (int*)(ws + A(2ull * MAXC * 4));
  int*   CNT = (int*)(ws + A(256));
  if (off > ws_size) return;

  // small-geometry padded buffers aliased into P768a (dead after conv1)
  size_t so = 0;
  auto SA = [&](size_t bytes) { size_t o = so; so = (so + bytes + 255) & ~(size_t)255; return o; };
  char* sb = (char*)P768a;
  _Float16* P384a = (_Float16*)(sb + SA(2ull * 386 * 386 * 64 * 2));
  _Float16* P384b = (_Float16*)(sb + SA(2ull * 386 * 386 * 64 * 2));
  _Float16* P192a = (_Float16*)(sb + SA(2ull * 194 * 194 * 128 * 2));
  _Float16* P192b = (_Float16*)(sb + SA(2ull * 194 * 194 * 128 * 2));
  _Float16* P96a  = (_Float16*)(sb + SA(2ull * 98 * 98 * 256 * 2));
  _Float16* P96b  = (_Float16*)(sb + SA(2ull * 98 * 98 * 256 * 2));
  _Float16* FT96  = (_Float16*)(sb + SA(2ull * 98 * 98 * 128 * 2));

  const float* image = (const float*)d_in[0];
  auto inp = [&](int j, int a) { return (const float*)d_in[1 + j * 6 + a]; }; // w,b,g,beta,mu,var

  // 1) pack weights (tap-major K) + fold BN params
  for (int j = 0; j < 12; ++j) {
    if (j > 0) {
      int n = cpad[j] * kpad[j];
      sp_pack_w<<<CDIV(n, 256), 256, 0, stream>>>(inp(j, 0), WGT + woff[j],
          BLKS[j].cout, BLKS[j].cin, BLKS[j].ksz * BLKS[j].ksz, kpad[j], n);
    }
    sp_prep_p<<<CDIV(cpad[j], 64), 64, 0, stream>>>(inp(j, 1), inp(j, 2), inp(j, 3),
                                                    inp(j, 4), inp(j, 5), PRM + poff[j],
                                                    BLKS[j].cout, cpad[j]);
  }

  // 2) image -> f16
  sp_cast_f16<<<CDIV((int)NS, 256), 256, 0, stream>>>(image, IMG, (int)NS);

  auto conv = [&](int j, const _Float16* X, _Float16* Y, int H, int W, int padout) {
    int HW = H * W;
    int lc = 31 - __builtin_clz((unsigned)BLKS[j].cin);
    dim3 grd(CDIV(HW, 16 * 8), cpad[j] / 16, 2);
    size_t shmem = (size_t)16 * kpad[j] * 2;
    if (BLKS[j].ksz == 3)
      sp_conv<3><<<grd, 256, shmem, stream>>>(X, WGT + woff[j], PRM + poff[j], Y,
                                              lc, H, W, cpad[j], kpad[j], BLKS[j].relu, padout);
    else
      sp_conv<1><<<grd, 256, shmem, stream>>>(X, WGT + woff[j], PRM + poff[j], Y,
                                              lc, H, W, cpad[j], kpad[j], BLKS[j].relu, padout);
  };
  auto pool = [&](const _Float16* X, _Float16* Y, int C, int H, int W) {
    int n = 2 * C * (H / 2) * (W / 2);
    sp_maxpool2<<<CDIV(n, 256), 256, 0, stream>>>(X, Y, H, W, C, n);
  };
  auto halo = [&](_Float16* X, int H, int W, int C) {
    int n = 2 * (2 * (W + 2) + 2 * H) * C;
    sp_zero_halo<<<CDIV(n, 256), 256, 0, stream>>>(X, H, W, C);
  };

  // 3) backbone (padded NHWC f16)
  halo(P768a, 768, 768, 64);
  sp_conv_c1<<<CDIV(2 * 768 * 768, 256), 256, 0, stream>>>(
      IMG, inp(0, 0), PRM + poff[0], P768a, 768, 768, 64);
  conv(1, P768a, P768b, 768, 768, 1);
  // P768a now dead; aliased small buffers come live below
  halo(P384a, 384, 384, 64);
  halo(P384b, 384, 384, 64);
  pool(P768b, P384a, 64, 768, 768);
  conv(2, P384a, P384b, 384, 384, 1);
  conv(3, P384b, P384a, 384, 384, 1);
  halo(P192a, 192, 192, 64);
  halo(P192b, 192, 192, 128);
  pool(P384a, P192a, 64, 384, 384);
  conv(4, P192a, P192b, 192, 192, 1);
  conv(5, P192b, P192a, 192, 192, 1);
  halo(P96a, 96, 96, 128);
  halo(P96b, 96, 96, 128);
  halo(FT96, 96, 96, 128);
  pool(P192a, P96a, 128, 192, 192);
  conv(6, P96a, P96b, 96, 96, 1);
  conv(7, P96b, FT96, 96, 96, 1);    // feat [2,98,98,128] padded

  // 4) heads (1x1 convs never read halos)
  conv(8, FT96, P96a, 96, 96, 1);    // [2,98,98,256] padded
  conv(9, P96a, DET, 96, 96, 0);     // [2,96,96,80] (65 valid)
  conv(10, FT96, P96a, 96, 96, 1);
  conv(11, P96a, DSC, 96, 96, 0);    // [2,96,96,256]

  // 5) softmax + pixel shuffle -> S [2,768,768]
  sp_softmax_shuffle<<<CDIV(2 * (int)hw96, 128), 128, 0, stream>>>(DET, S, 96, 96);

  // 6) NMS (radius 4, separable 9x9 max)
  const int n2 = (int)NS, gb = CDIV(n2, 256);
  sp_winmax_h<<<gb, 256, 0, stream>>>(S, T3, 768, n2, 4);
  sp_winmax_v<<<gb, 256, 0, stream>>>(T3, T1, 768, 768, n2, 4);
  sp_eqmask<<<gb, 256, 0, stream>>>(S, T1, Mm, n2);
  for (int it = 0; it < 2; ++it) {
    sp_winmax_h<<<gb, 256, 0, stream>>>(Mm, T3, 768, n2, 4);
    sp_winmax_v<<<gb, 256, 0, stream>>>(T3, T1, 768, 768, n2, 4);   // T1 = mp(mask)
    sp_suppress<<<gb, 256, 0, stream>>>(T1, S, T2, n2);             // supp_scores
    sp_winmax_h<<<gb, 256, 0, stream>>>(T2, T3, 768, n2, 4);
    sp_winmax_v<<<gb, 256, 0, stream>>>(T3, T4, 768, 768, n2, 4);   // mp(supp_scores)
    sp_mask_update<<<gb, 256, 0, stream>>>(Mm, T1, T2, T4, n2);
  }
  sp_finalize<<<gb, 256, 0, stream>>>(Mm, S, SF, 768, 768);

  // 7) top-K: compact + deterministic bitonic sort + emit kps/scores
  sp_init_cand<<<CDIV(2 * MAXC, 256), 256, 0, stream>>>(CV, CI, CNT, 2 * MAXC);
  sp_compact<<<gb, 256, 0, stream>>>(SF, CV, CI, CNT, (int)(NS / 2));
  sp_bitonic<<<2, 256, 0, stream>>>(CV, CI);
  float* out = (float*)d_out;
  sp_emit<<<CDIV(2 * KP, 256), 256, 0, stream>>>(CV, CI, out, 768);

  // 8) descriptor normalize + bilinear sample at keypoints
  sp_desc_norm<<<CDIV(2 * (int)hw96, 128), 128, 0, stream>>>(DSC, DN, 2 * (int)hw96);
  sp_sample<<<2 * KP, 256, 0, stream>>>(DN, out, out + 2 * KP * 2 + 2 * KP);
}
